// AttentionEncoder_50835232915591
// MI455X (gfx1250) — matrix operand order repaired and checked by reading, not compiled, not run on hardware
//
#include <hip/hip_runtime.h>
#include <hip/hip_bf16.h>

typedef __attribute__((ext_vector_type(16))) __bf16 v16bf;
typedef __attribute__((ext_vector_type(8)))  __bf16 v8bf;
typedef __attribute__((ext_vector_type(8)))  float  v8f;

#define B_   8
#define C_   3
#define S_   1024
#define DIM_ 768
#define H_   3
#define HD_  256
#define BCH_ (B_*C_*H_)                 // 72
#define PER_ ((size_t)BCH_*S_*HD_)      // elems per Q/K/Vt tensor

// ---------------------------------------------------------------------------
// Kernel 1: per-head QKV projection, fp32 -> bf16, via v_wmma_f32_16x16x32_bf16
//   Q[bch][s][e] = sum_d X[b][c][s][h*HD+d] * W[h][e][d] + b[h][e]
//   Q,K stored row-major [S][HD]; V stored transposed [HD][S] for PV B-frags.
// ---------------------------------------------------------------------------
__global__ __launch_bounds__(128)
void qkv_proj_kernel(const float* __restrict__ x,
                     const float* __restrict__ Wq, const float* __restrict__ bq,
                     const float* __restrict__ Wk, const float* __restrict__ bk,
                     const float* __restrict__ Wv, const float* __restrict__ bv,
                     __bf16* __restrict__ qg, __bf16* __restrict__ kg,
                     __bf16* __restrict__ vtg)
{
    const int lane = threadIdx.x & 31;
    const int wave = threadIdx.x >> 5;
    const int task = blockIdx.x * 4 + wave;       // 72*64*16 tasks
    const int et   = task & 15;
    const int st   = (task >> 4) & 63;
    const int bch  = task >> 10;
    const int h    = bch % H_;
    const int bc   = bch / H_;

    const int row  = lane & 15;
    const int half = lane >> 4;

    const int s_a = st * 16 + row;                // A-frag row (s)
    const int e_b = et * 16 + row;                // B-frag col (e)

    const float* xrow = x + ((size_t)(bc * S_ + s_a)) * DIM_ + h * HD_;
    const float* wqh  = Wq + (size_t)h * HD_ * HD_ + (size_t)e_b * HD_;
    const float* wkh  = Wk + (size_t)h * HD_ * HD_ + (size_t)e_b * HD_;
    const float* wvh  = Wv + (size_t)h * HD_ * HD_ + (size_t)e_b * HD_;

    v8f cq = {}, ck = {}, cv = {};
    #pragma unroll
    for (int kk = 0; kk < 8; ++kk) {              // K = 256 in chunks of 32
        const int d0 = kk * 32;
        v16bf a;                                   // A layout: K=half*16+e (contig d)
        #pragma unroll
        for (int e = 0; e < 8; ++e) a[e]     = (__bf16)xrow[d0 + half * 16 + e];
        #pragma unroll
        for (int e = 0; e < 8; ++e) a[8 + e] = (__bf16)xrow[d0 + half * 16 + 8 + e];
        v16bf bqv, bkv, bvv;                       // B layout: K=half*16+e (contig d)
        #pragma unroll
        for (int e = 0; e < 16; ++e) {
            bqv[e] = (__bf16)wqh[d0 + half * 16 + e];
            bkv[e] = (__bf16)wkh[d0 + half * 16 + e];
            bvv[e] = (__bf16)wvh[d0 + half * 16 + e];
        }
        cq = __builtin_amdgcn_wmma_f32_16x16x32_bf16(false, a, false, bqv, (short)0, cq, false, false);
        ck = __builtin_amdgcn_wmma_f32_16x16x32_bf16(false, a, false, bkv, (short)0, ck, false, false);
        cv = __builtin_amdgcn_wmma_f32_16x16x32_bf16(false, a, false, bvv, (short)0, cv, false, false);
    }

    const float biasq = bq[h * HD_ + e_b];
    const float biask = bk[h * HD_ + e_b];
    const float biasv = bv[h * HD_ + e_b];

    #pragma unroll
    for (int r = 0; r < 8; ++r) {                 // C/D: M = r + 8*half, N = lane%16
        const int s = st * 16 + r + 8 * half;
        qg [((size_t)bch * S_  + s)   * HD_ + e_b] = (__bf16)(cq[r] + biasq);
        kg [((size_t)bch * S_  + s)   * HD_ + e_b] = (__bf16)(ck[r] + biask);
        vtg[((size_t)bch * HD_ + e_b) * S_  + s  ] = (__bf16)(cv[r] + biasv);
    }
}

// ---------------------------------------------------------------------------
// Kernel 2: fused flash attention per (b,c,h). 8 waves/block, each wave owns
// a 16-row query tile; K/V 32-key tiles are staged once per block into LDS
// with GLOBAL_LOAD_ASYNC_TO_LDS_B128 (ASYNCcnt) and shared by all 8 waves.
// ---------------------------------------------------------------------------
__global__ __launch_bounds__(256)
void flash_attn_kernel(const __bf16* __restrict__ qg,
                       const __bf16* __restrict__ kg,
                       const __bf16* __restrict__ vtg,
                       float* __restrict__ out)
{
    // K tile: 32 keys x 256 d, padded pitch 264 bf16 (528B, 16B-multiple)
    __shared__ __align__(16) __bf16 sK[32][264];
    // V tile: 256 n x 32 keys, padded pitch 40 bf16 (80B, 16B-multiple)
    __shared__ __align__(16) __bf16 sV[256][40];
    __shared__ float  sS[8][16][33];    // per-wave score tile
    __shared__ __bf16 sP[8][16][40];    // per-wave softmaxed probs
    __shared__ float  sAux[8][16];      // per-wave row corr / l broadcast

    const int tid  = threadIdx.x;
    const int lane = tid & 31;
    const int w    = tid >> 5;
    const int bch  = blockIdx.x >> 3;
    const int qt   = blockIdx.x & 7;
    const int h    = bch % H_;
    const int bc   = bch / H_;
    const int b    = bc / C_;
    const int c    = bc % C_;

    const int row  = lane & 15;
    const int half = lane >> 4;
    const int qs0  = qt * 128 + w * 16;

    const __bf16* qrow = qg + ((size_t)bch * S_ + qs0 + row) * HD_;
    const unsigned long long kt_base =
        (unsigned long long)(const void*)(kg + ((size_t)bch * S_) * HD_);
    const unsigned long long vt_base =
        (unsigned long long)(const void*)(vtg + (size_t)bch * HD_ * S_);

    v8f o[16];
    #pragma unroll
    for (int t = 0; t < 16; ++t) o[t] = (v8f){};
    float m_run = -1e30f;
    float l_run = 0.f;

    for (int kt = 0; kt < 32; ++kt) {
        const int key0 = kt * 32;

        // ---- cooperative async stage of K (16KB) and V (16KB) tiles ----
        // K: 32 rows x 32 16B-chunks = 1024 chunks; 4 per thread
        #pragma unroll
        for (int i = 0; i < 4; ++i) {
            const int idx = tid * 4 + i;
            const int kr  = idx >> 5;
            const int ch  = idx & 31;
            const unsigned lds_off =
                (unsigned)(size_t)(&sK[kr][0]) + (unsigned)(ch * 16);
            const unsigned long long ga =
                kt_base + ((unsigned long long)(key0 + kr) * HD_ + 0ull) * 2ull
                        + (unsigned long long)(ch * 16);
            asm volatile("global_load_async_to_lds_b128 %0, %1, off"
                         :: "v"(lds_off), "v"(ga) : "memory");
        }
        // V: 256 rows x 4 16B-chunks = 1024 chunks; 4 per thread
        #pragma unroll
        for (int i = 0; i < 4; ++i) {
            const int idx = tid * 4 + i;
            const int nr  = idx >> 2;
            const int ch  = idx & 3;
            const unsigned lds_off =
                (unsigned)(size_t)(&sV[nr][0]) + (unsigned)(ch * 16);
            const unsigned long long ga =
                vt_base + ((unsigned long long)nr * S_ + (unsigned long long)key0) * 2ull
                        + (unsigned long long)(ch * 16);
            asm volatile("global_load_async_to_lds_b128 %0, %1, off"
                         :: "v"(lds_off), "v"(ga) : "memory");
        }
        asm volatile("s_wait_asynccnt 0x0" ::: "memory");
        __syncthreads();

        // ---- S = (Q K^T) * scale : two 16x16 C-frags over K=256 ----
        v8f c0 = {}, c1 = {};
        #pragma unroll
        for (int kk = 0; kk < 8; ++kk) {
            const int d0 = kk * 32;
            v16bf a;
            *(v8bf*)&a       = *(const v8bf*)(qrow + d0 + half * 16);
            *((v8bf*)&a + 1) = *(const v8bf*)(qrow + d0 + half * 16 + 8);
            v16bf b0 = *(const v16bf*)(&sK[row][d0 + half * 16]);
            v16bf b1 = *(const v16bf*)(&sK[16 + row][d0 + half * 16]);
            c0 = __builtin_amdgcn_wmma_f32_16x16x32_bf16(false, a, false, b0, (short)0, c0, false, false);
            c1 = __builtin_amdgcn_wmma_f32_16x16x32_bf16(false, a, false, b1, (short)0, c1, false, false);
        }
        #pragma unroll
        for (int r = 0; r < 8; ++r) {
            sS[w][r + 8 * half][row]      = c0[r] * 0.0625f;   // 1/sqrt(256)
            sS[w][r + 8 * half][16 + row] = c1[r] * 0.0625f;
        }

        // ---- online softmax: lane handles row lane%16, cols half*16.. ----
        float tmax = -1e30f;
        #pragma unroll
        for (int j = 0; j < 16; ++j) tmax = fmaxf(tmax, sS[w][row][half * 16 + j]);
        tmax = fmaxf(tmax, __shfl_xor(tmax, 16, 32));
        const float m_new = fmaxf(m_run, tmax);
        const float corr  = __expf(m_run - m_new);
        float psum = 0.f;
        #pragma unroll
        for (int j = 0; j < 16; ++j) {
            const float p = __expf(sS[w][row][half * 16 + j] - m_new);
            sP[w][row][half * 16 + j] = (__bf16)p;
            psum += p;
        }
        psum += __shfl_xor(psum, 16, 32);
        l_run = l_run * corr + psum;
        m_run = m_new;
        sAux[w][row] = corr;

        // ---- rescale accumulators ----
        float cr[8];
        #pragma unroll
        for (int r = 0; r < 8; ++r) cr[r] = sAux[w][r + 8 * half];
        #pragma unroll
        for (int t = 0; t < 16; ++t)
            #pragma unroll
            for (int r = 0; r < 8; ++r) o[t][r] *= cr[r];

        // ---- O += P V : P A-frag from LDS, V B-frags from staged LDS ----
        v16bf pfrag;
        *(v8bf*)&pfrag       = *(const v8bf*)(&sP[w][row][half * 16]);
        *((v8bf*)&pfrag + 1) = *(const v8bf*)(&sP[w][row][half * 16 + 8]);
        #pragma unroll
        for (int t = 0; t < 16; ++t) {
            v16bf bvf = *(const v16bf*)(&sV[t * 16 + row][half * 16]);
            o[t] = __builtin_amdgcn_wmma_f32_16x16x32_bf16(false, pfrag, false, bvf, (short)0, o[t], false, false);
        }

        __syncthreads();   // all waves done with sK/sV before next staging
    }

    // ---- finalize: divide by l, write out[b][s][c*DIM + h*HD + n] ----
    sAux[w][row] = l_run;
    float il[8];
    #pragma unroll
    for (int r = 0; r < 8; ++r) il[r] = 1.f / sAux[w][r + 8 * half];

    #pragma unroll
    for (int t = 0; t < 16; ++t) {
        #pragma unroll
        for (int r = 0; r < 8; ++r) {
            const int s = qs0 + r + 8 * half;
            out[((size_t)b * S_ + s) * (C_ * DIM_) + c * DIM_ + h * HD_ + t * 16 + row]
                = o[t][r] * il[r];
        }
    }
}

// ---------------------------------------------------------------------------
extern "C" void kernel_launch(void* const* d_in, const int* in_sizes, int n_in,
                              void* d_out, int out_size, void* d_ws, size_t ws_size,
                              hipStream_t stream)
{
    (void)in_sizes; (void)n_in; (void)out_size;
    const float* x  = (const float*)d_in[0];
    const float* Wq = (const float*)d_in[1];
    const float* bq = (const float*)d_in[2];
    const float* Wk = (const float*)d_in[3];
    const float* bk = (const float*)d_in[4];
    const float* Wv = (const float*)d_in[5];
    const float* bv = (const float*)d_in[6];
    float* out = (float*)d_out;

    if (ws_size < 3 * PER_ * sizeof(__bf16)) return;   // need ~113 MB scratch
    __bf16* qg  = (__bf16*)d_ws;
    __bf16* kg  = qg + PER_;
    __bf16* vtg = kg + PER_;

    qkv_proj_kernel<<<BCH_ * 64 * 16 / 4, 128, 0, stream>>>(
        x, Wq, bq, Wk, bk, Wv, bv, qg, kg, vtg);

    flash_attn_kernel<<<BCH_ * 8, 256, 0, stream>>>(qg, kg, vtg, out);
}
